// RoutedAllFC_66941360276016
// MI455X (gfx1250) — matrix operand order repaired and checked by reading, not compiled, not run on hardware
//
#include <hip/hip_runtime.h>
#include <hip/hip_bf16.h>

// ---------------------------------------------------------------------------
// RoutedAllFC on MI455X (gfx1250, wave32, WMMA 16x16x32 f16)
//   conv1     : NCHW f32 input, implicit-GEMM WMMA (K=27->32), channel-last out
//   conv2..4  : channel-last [H][W][32] f16; 3x3 conv = 9 shifted K=32 GEMMs,
//               A-fragments are contiguous 64B runs (2 aligned v8h LDS loads),
//               input staged with GLOBAL_LOAD_ASYNC_TO_LDS_B128 (ASYNCcnt)
//   bn_stats  : per-channel mean / rsqrt(var+eps) over conv4 output
//   routing   : BN-apply + argmax routing + 3 gathered expert GEMVs (VALU)
// ---------------------------------------------------------------------------

typedef __attribute__((ext_vector_type(16))) _Float16 v16h;
typedef __attribute__((ext_vector_type(8)))  _Float16 v8h;
typedef __attribute__((ext_vector_type(8)))  float    v8f;
typedef __attribute__((ext_vector_type(4)))  int      v4i;

#define DEVINL __device__ __forceinline__

constexpr int BATCH = 4096;

#if __has_builtin(__builtin_amdgcn_global_load_async_to_lds_b128) && \
    __has_builtin(__builtin_amdgcn_s_wait_asynccnt)
#define HAVE_ASYNC_LDS 1
typedef __attribute__((address_space(1))) v4i* gv4ip;   // global int4*
typedef __attribute__((address_space(3))) v4i* lv4ip;   // LDS int4*
#else
#define HAVE_ASYNC_LDS 0
#endif

// max with the value from lane^16 (cross-half), no LDS traffic
DEVINL float maxx16(float v) {
#if __has_builtin(__builtin_amdgcn_permlanex16)
  const int s = __float_as_int(v);
  const int o = __builtin_amdgcn_permlanex16(s, s, 0x76543210, 0xfedcba98, false, false);
  return fmaxf(v, __int_as_float(o));
#else
  return fmaxf(v, __shfl_xor(v, 16));
#endif
}

// A-fragment from a contiguous 32-half (64B) run at p (ISA A layout: lane
// half h holds K {8h..8h+7, 16+8h..16+8h+7})
DEVINL v16h load_fragA_ptr(const _Float16* p, int h) {
  v8h lo = *(const v8h*)(p + h * 8);
  v8h hi = *(const v8h*)(p + 16 + h * 8);
  return __builtin_shufflevector(lo, hi, 0,1,2,3,4,5,6,7,8,9,10,11,12,13,14,15);
}

// B-fragment (32x16 f16), stored as B^T rows [N][K=32] in LDS:
// col N=lane%16; lane half h holds K {8h..8h+7, 16+8h..16+8h+7}.
DEVINL v16h load_fragB(const _Float16* base, int stride, int k0, int lane) {
  const int n = lane & 15;
  const int h = lane >> 4;
  v8h lo = *(const v8h*)(base + n * stride + k0 + h * 8);
  v8h hi = *(const v8h*)(base + n * stride + k0 + 16 + h * 8);
  return __builtin_shufflevector(lo, hi, 0,1,2,3,4,5,6,7,8,9,10,11,12,13,14,15);
}

// ---------------------------------------------------------------------------
// conv1: NCHW f32 x[B,3,32,32] -> channel-last f16 [16][16][32].
// M-tile = 2x8 output pixels; K = 27 padded to 32 (single WMMA K-step).
// ---------------------------------------------------------------------------
__global__ __launch_bounds__(128, 1)
void conv1_kernel(const float* __restrict__ in, const float* __restrict__ w,
                  const float* __restrict__ bias, _Float16* __restrict__ out) {
  constexpr int CIN = 3, H = 32, K = 27, KP = 32;
  constexpr int SP = 40, CSTART = 8;        // padded NCHW LDS image row stride
  constexpr int TW = 8, TILESX = H / TW, NTILES = (H / 2) * TILESX;
  constexpr int HO = H / 2;

  __shared__ alignas(32) _Float16 sIn[CIN * (H + 2) * SP];
  __shared__ alignas(32) _Float16 sW[32 * KP];   // weights as B^T [O][K]

  const int tid  = threadIdx.x;
  const int lane = tid & 31;
  const int wv   = tid >> 5;
  const int b    = blockIdx.x;

  for (int idx = tid; idx < CIN * (H + 2) * SP; idx += 128) {
    const int ci = idx / ((H + 2) * SP);
    const int r  = idx % ((H + 2) * SP);
    const int yy = r / SP, cc = r % SP;
    const int py = yy - 1, px = cc - CSTART;
    _Float16 v = (_Float16)0.f;
    if (py >= 0 && py < H && px >= 0 && px < H)
      v = (_Float16)in[((size_t)(b * CIN + ci) * H + py) * H + px];
    sIn[idx] = v;
  }
  for (int idx = tid; idx < 32 * KP; idx += 128) {
    const int o = idx / KP, k = idx % KP;
    sW[idx] = (k < K) ? (_Float16)w[o * K + k] : (_Float16)0.f;
  }
  __syncthreads();

  for (int t = wv; t < NTILES; t += 4) {
    const int y0 = (t / TILESX) * 2;
    const int x0 = (t % TILESX) * TW;
    const int r  = lane & 15;
    const int py = y0 + r / TW, px = x0 + r % TW;
    const int base = py * SP + px + (CSTART - 1);
    const bool lohalf = (lane < 16);

    v16h afr = {};
    #pragma unroll
    for (int j = 0; j < 16; ++j) {          // register-direct im2col gather
      const int klo = j + (j >= 8 ? 8 : 0);
      const int khi = klo + 8;
      const int olo = (klo / 9) * (H + 2) * SP + ((klo % 9) / 3) * SP + (klo % 9) % 3;
      const int ohi = (khi < K)
          ? ((khi / 9) * (H + 2) * SP + ((khi % 9) / 3) * SP + (khi % 9) % 3) : 0;
      _Float16 v = sIn[base + (lohalf ? olo : ohi)];
      if (khi >= K) v = lohalf ? v : (_Float16)0.f;   // K=27 tail
      afr[j] = v;
    }

    #pragma unroll
    for (int nt = 0; nt < 2; ++nt) {
      v16h bf = load_fragB(sW + nt * 16 * KP, KP, 0, lane);
      v8f acc = {};
      acc = __builtin_amdgcn_wmma_f32_16x16x32_f16(
          false, afr, false, bf, (short)0, acc, false, false);
      const int   o  = nt * 16 + (lane & 15);
      const float bv = bias[o];
      #pragma unroll
      for (int i = 0; i < 8; ++i) {
        const float v = acc[i] + bv;
        acc[i] = v > 0.f ? v : 0.f;
      }
      #pragma unroll
      for (int c = 0; c < 4; ++c) {         // 2x2 maxpool, channel-last store
        const float pv = maxx16(fmaxf(acc[2 * c], acc[2 * c + 1]));
        if (lane < 16)
          out[(size_t)b * HO * HO * 32 +
              (((y0 >> 1) * HO) + (x0 >> 1) + c) * 32 + o] = (_Float16)pv;
      }
    }
  }
}

// ---------------------------------------------------------------------------
// Channel-last conv block (Cin=Cout=32). Input [H][W][32] f16, padded LDS
// image [(H+2)][(W+2)][32] with zero border pixels. 3x3 conv = 9 K=32 GEMMs.
// ---------------------------------------------------------------------------
template <int HIN, int TH, int NWAVES, typename OutT>
__global__ __launch_bounds__(NWAVES * 32, 1)
void convcl_kernel(const _Float16* __restrict__ in, const float* __restrict__ w,
                   const float* __restrict__ bias, OutT* __restrict__ out) {
  constexpr int H  = HIN;
  constexpr int PW = H + 2;                 // padded width (pixels)
  constexpr int TW = 16 / TH;
  constexpr int TILESX = H / TW;
  constexpr int NTILES = (H / TH) * TILESX;
  constexpr int HO = H / 2;
  constexpr int NTHR = NWAVES * 32;

  __shared__ alignas(32) _Float16 sIn[PW * PW * 32];
  __shared__ alignas(32) _Float16 sW[9 * 32 * 32];  // [t=dy*3+dx][o][ci]

  const int tid  = threadIdx.x;
  const int lane = tid & 31;
  const int wv   = tid >> 5;
  const int b    = blockIdx.x;

  // ---- zero the 1-pixel border (8 halves per unit, disjoint from interior)
  constexpr int NBORD = 2 * PW + 2 * H;     // border pixels
  for (int i = tid; i < NBORD * 4; i += NTHR) {
    const int p = i >> 2, part = i & 3;
    int yy, xx;
    if (p < PW)            { yy = 0;          xx = p; }
    else if (p < 2 * PW)   { yy = H + 1;      xx = p - PW; }
    else if (p < 2 * PW + H) { yy = p - 2 * PW + 1; xx = 0; }
    else                   { yy = p - 2 * PW - H + 1; xx = H + 1; }
    *(v8h*)&sIn[(yy * PW + xx) * 32 + part * 8] = (v8h)(_Float16)0.f;
  }
  // ---- interior: async raw copy (rows are H*64 contiguous bytes) ----------
#if HAVE_ASYNC_LDS
  {
    constexpr int XPR = H * 4;              // b128 transfers per row
    for (int i = tid; i < H * XPR; i += NTHR) {
      const int y = i / XPR, part = i % XPR;
      const _Float16* g = in + (size_t)b * H * H * 32 + (size_t)y * H * 32 + part * 8;
      _Float16*       l = &sIn[((y + 1) * PW + 1) * 32 + part * 8];
      __builtin_amdgcn_global_load_async_to_lds_b128((gv4ip)g, (lv4ip)l, 0, 0);
    }
    __builtin_amdgcn_s_wait_asynccnt(0);
  }
#else
  for (int i = tid; i < H * H * 32 / 8; i += NTHR) {
    const int y = i / (H * 4), part = i % (H * 4);
    v8h v = *(const v8h*)(in + (size_t)b * H * H * 32 + (size_t)y * H * 32 + part * 8);
    *(v8h*)&sIn[((y + 1) * PW + 1) * 32 + part * 8] = v;
  }
#endif
  // ---- weights: [O][Cin][3][3] f32 -> [t][o][ci] f16 ----------------------
  for (int idx = tid; idx < 9 * 32 * 32; idx += NTHR) {
    const int t = idx >> 10, o = (idx >> 5) & 31, ci = idx & 31;
    sW[idx] = (_Float16)w[o * 288 + ci * 9 + t];
  }
  __syncthreads();

  // ---- tile loop (no LDS writes inside -> no barriers) --------------------
  for (int t = wv; t < NTILES; t += NWAVES) {
    const int y0 = (t / TILESX) * TH;
    const int x0 = (t % TILESX) * TW;
    const int r  = lane & 15;
    const int py = y0 + r / TW, px = x0 + r % TW;
    const int h  = lane >> 4;
    const _Float16* pbase = sIn + (py * PW + px) * 32;   // patch corner pixel

    v16h afr[9];
    #pragma unroll
    for (int s = 0; s < 9; ++s)             // contiguous 64B run per (dy,dx)
      afr[s] = load_fragA_ptr(pbase + ((s / 3) * PW + (s % 3)) * 32, h);

    #pragma unroll
    for (int nt = 0; nt < 2; ++nt) {
      v8f acc = {};
      #pragma unroll
      for (int s = 0; s < 9; ++s) {
        v16h bf = load_fragB(sW + s * 1024 + nt * 16 * 32, 32, 0, lane);
        acc = __builtin_amdgcn_wmma_f32_16x16x32_f16(
            false, afr[s], false, bf, (short)0, acc, false, false);
      }
      const int   o  = nt * 16 + (lane & 15);
      const float bv = bias[o];
      #pragma unroll
      for (int i = 0; i < 8; ++i) {
        const float v = acc[i] + bv;
        acc[i] = v > 0.f ? v : 0.f;
      }
      // fused 2x2 maxpool, channel-last store (lanes = consecutive channels)
      if constexpr (TH == 2) {              // m=(y-y0)*8+(x-x0)
        #pragma unroll
        for (int c = 0; c < 4; ++c) {
          const float pv = maxx16(fmaxf(acc[2 * c], acc[2 * c + 1]));
          if (lane < 16)
            out[(size_t)b * HO * HO * 32 +
                (((y0 >> 1) * HO) + (x0 >> 1) + c) * 32 + o] = (OutT)pv;
        }
      } else {                              // TH==4: m=(y-y0)*4+(x-x0)
        const int hh = lane >> 4;
        #pragma unroll
        for (int c = 0; c < 2; ++c) {
          const float pv = fmaxf(fmaxf(acc[2 * c], acc[2 * c + 1]),
                                 fmaxf(acc[2 * c + 4], acc[2 * c + 5]));
          out[(size_t)b * HO * HO * 32 +
              ((((y0 >> 1) + hh) * HO) + (x0 >> 1) + c) * 32 + o] = (OutT)pv;
        }
      }
    }
  }
}

// ---------------------------------------------------------------------------
// BN batch statistics over conv4 output, channel-last [B][2][2][32] f32.
// stats[c] = mean, stats[32+c] = rsqrt(var + eps)
// ---------------------------------------------------------------------------
__global__ __launch_bounds__(256)
void bn_stats_kernel(const float* __restrict__ act4, float* __restrict__ stats) {
  const int c = blockIdx.x;
  const int tid = threadIdx.x;
  float s = 0.f, s2 = 0.f;
  for (int b = tid; b < BATCH; b += 256) {
    const float* p = act4 + (size_t)b * 128 + c;
    #pragma unroll
    for (int i = 0; i < 4; ++i) { const float v = p[i * 32]; s += v; s2 += v * v; }
  }
  __shared__ float rs[256], rq[256];
  rs[tid] = s; rq[tid] = s2;
  __syncthreads();
  for (int st = 128; st > 0; st >>= 1) {
    if (tid < st) { rs[tid] += rs[tid + st]; rq[tid] += rq[tid + st]; }
    __syncthreads();
  }
  if (tid == 0) {
    const float n  = (float)(BATCH * 4);
    const float mu = rs[0] / n;
    const float vr = rq[0] / n - mu * mu;
    stats[c]      = mu;
    stats[32 + c] = rsqrtf(vr + 1e-5f);
  }
}

// Cross-lane argmax with first-index tie-break (matches jnp.argmax)
DEVINL void argmax32(float& v, int& idx) {
  #pragma unroll
  for (int m = 16; m >= 1; m >>= 1) {
    const float ov = __shfl_xor(v, m);
    const int   oi = __shfl_xor(idx, m);
    if (ov > v || (ov == v && oi < idx)) { v = ov; idx = oi; }
  }
}

// ---------------------------------------------------------------------------
// BN-apply + routed MoE head. One wave per sample (8 samples / 256-thr block).
// feat order j = c*4 + y*2 + x maps to channel-last src (y*2+x)*32 + c.
// ---------------------------------------------------------------------------
__global__ __launch_bounds__(256)
void routing_kernel(const float* __restrict__ act4, const float* __restrict__ stats,
                    const float* __restrict__ bn_g, const float* __restrict__ bn_b,
                    const float* __restrict__ wpt, const float* __restrict__ wd1,
                    const float* __restrict__ wd2, const float* __restrict__ wd3,
                    const float* __restrict__ s1w, const float* __restrict__ s1b,
                    const float* __restrict__ s2w, const float* __restrict__ s2b,
                    const float* __restrict__ s3w, const float* __restrict__ s3b,
                    float* __restrict__ out) {
  __shared__ float fbuf[8][128];
  __shared__ float h1[8][48];
  __shared__ float h2[8][48];
  const int lane = threadIdx.x & 31;
  const int wv   = threadIdx.x >> 5;
  const int s    = blockIdx.x * 8 + wv;

  for (int j = lane; j < 128; j += 32) {
    const int c = j >> 2, yx = j & 3;
    const float v = act4[(size_t)s * 128 + yx * 32 + c];
    fbuf[wv][j] = (v - stats[c]) * stats[32 + c] * bn_g[c] + bn_b[c];
  }
  __syncthreads();

  // agent = argmax(feat @ wpt)  [10]
  float v = -__builtin_inff(); int idx = 0x7fffffff;
  if (lane < 10) {
    float a = 0.f;
    for (int f = 0; f < 128; ++f) a += fbuf[wv][f] * wpt[f * 10 + lane];
    v = a; idx = lane;
  }
  argmax32(v, idx);
  const int agent = idx;

  // a1 = argmax(feat @ wd1[agent])  [16]
  v = -__builtin_inff(); idx = 0x7fffffff;
  if (lane < 16) {
    float a = 0.f;
    const float* wp = wd1 + ((size_t)agent * 128) * 16 + lane;
    for (int f = 0; f < 128; ++f) a += fbuf[wv][f] * wp[f * 16];
    v = a; idx = lane;
  }
  argmax32(v, idx);
  const int a1 = idx;

  // h1 = relu(feat @ s1w[a1] + s1b[a1])  [48]
  for (int o = lane; o < 48; o += 32) {
    float a = s1b[a1 * 48 + o];
    const float* wp = s1w + ((size_t)a1 * 128) * 48 + o;
    for (int f = 0; f < 128; ++f) a += fbuf[wv][f] * wp[f * 48];
    h1[wv][o] = a > 0.f ? a : 0.f;
  }
  __syncthreads();

  // a2
  v = -__builtin_inff(); idx = 0x7fffffff;
  if (lane < 16) {
    float a = 0.f;
    const float* wp = wd2 + ((size_t)agent * 48) * 16 + lane;
    for (int f = 0; f < 48; ++f) a += h1[wv][f] * wp[f * 16];
    v = a; idx = lane;
  }
  argmax32(v, idx);
  const int a2 = idx;

  // h2 = relu(h1 @ s2w[a2] + s2b[a2])  [48]
  for (int o = lane; o < 48; o += 32) {
    float a = s2b[a2 * 48 + o];
    const float* wp = s2w + ((size_t)a2 * 48) * 48 + o;
    for (int f = 0; f < 48; ++f) a += h1[wv][f] * wp[f * 48];
    h2[wv][o] = a > 0.f ? a : 0.f;
  }
  __syncthreads();

  // a3
  v = -__builtin_inff(); idx = 0x7fffffff;
  if (lane < 16) {
    float a = 0.f;
    const float* wp = wd3 + ((size_t)agent * 48) * 16 + lane;
    for (int f = 0; f < 48; ++f) a += h2[wv][f] * wp[f * 16];
    v = a; idx = lane;
  }
  argmax32(v, idx);
  const int a3 = idx;

  // out = h2 @ s3w[a3] + s3b[a3]  [10]
  if (lane < 10) {
    float a = s3b[a3 * 10 + lane];
    const float* wp = s3w + ((size_t)a3 * 48) * 10 + lane;
    for (int f = 0; f < 48; ++f) a += h2[wv][f] * wp[f * 10];
    out[(size_t)s * 10 + lane] = a;
  }
}

// ---------------------------------------------------------------------------
extern "C" void kernel_launch(void* const* d_in, const int* in_sizes, int n_in,
                              void* d_out, int out_size, void* d_ws, size_t ws_size,
                              hipStream_t stream) {
  const float* x   = (const float*)d_in[0];
  // d_in[1] = tasks (unused by the reference math)
  const float* cw1 = (const float*)d_in[2];
  const float* cb1 = (const float*)d_in[3];
  const float* cw2 = (const float*)d_in[4];
  const float* cb2 = (const float*)d_in[5];
  const float* cw3 = (const float*)d_in[6];
  const float* cb3 = (const float*)d_in[7];
  const float* cw4 = (const float*)d_in[8];
  const float* cb4 = (const float*)d_in[9];
  const float* bng = (const float*)d_in[10];
  const float* bnb = (const float*)d_in[11];
  const float* wpt = (const float*)d_in[12];
  const float* wd1 = (const float*)d_in[13];
  const float* wd2 = (const float*)d_in[14];
  const float* wd3 = (const float*)d_in[15];
  const float* s1w = (const float*)d_in[16];
  const float* s1b = (const float*)d_in[17];
  const float* s2w = (const float*)d_in[18];
  const float* s2b = (const float*)d_in[19];
  const float* s3w = (const float*)d_in[20];
  const float* s3b = (const float*)d_in[21];
  float* outp = (float*)d_out;

  // workspace: channel-last f16 activations, f32 conv4 output + stats
  const size_t n1 = (size_t)BATCH * 16 * 16 * 32;   // act1
  const size_t n2 = (size_t)BATCH * 8 * 8 * 32;     // act2
  const size_t n3 = (size_t)BATCH * 4 * 4 * 32;     // act3
  _Float16* act1 = (_Float16*)d_ws;
  _Float16* act2 = act1 + n1;
  _Float16* act3 = act2 + n2;
  float*    act4 = (float*)(act3 + n3);             // BATCH*128 f32 (2x2x32)
  float*    stats = act4 + (size_t)BATCH * 128;     // 64 f32

  conv1_kernel<<<BATCH, 128, 0, stream>>>(x, cw1, cb1, act1);
  convcl_kernel<16, 2, 4, _Float16><<<BATCH, 128, 0, stream>>>(act1, cw2, cb2, act2);
  convcl_kernel<8, 2, 4, _Float16><<<BATCH, 128, 0, stream>>>(act2, cw3, cb3, act3);
  convcl_kernel<4, 4, 1, float><<<BATCH, 32, 0, stream>>>(act3, cw4, cb4, act4);

  bn_stats_kernel<<<32, 256, 0, stream>>>(act4, stats);

  routing_kernel<<<BATCH / 8, 256, 0, stream>>>(
      act4, stats, bng, bnb, wpt, wd1, wd2, wd3,
      s1w, s1b, s2w, s2b, s3w, s3b, outp);
}
